// GemmaAttention_43980465111061
// MI455X (gfx1250) — compile-verified, matrix operand order checked by reading, not run on hardware
//
#include <hip/hip_runtime.h>
#include <hip/hip_bf16.h>

// Problem constants (Gemma-style attention, from reference)
#define HDIM   2048
#define NHEADS 8
#define HEADD  256
#define BATCH  2
#define SEQ    2048
#define MROWS  (BATCH * SEQ)       // 4096
#define SCALE_Q 0.0625f            // 256^-0.5
#define NEGMASK -1e9f
#define LN10000 9.210340371976184f

// GEMM tiling: block = 256 threads = 8 waves; macro tile 64(M) x 128(N), K-step 32
#define BM 64
#define BN 128
#define BK 32

typedef __attribute__((ext_vector_type(16))) __bf16 bf16x16;
typedef __attribute__((ext_vector_type(8)))  __bf16 bf16x8;
typedef __attribute__((ext_vector_type(2)))  __bf16 bf16x2;
typedef __attribute__((ext_vector_type(8)))  float  f32x8;
typedef __attribute__((ext_vector_type(4)))  float  f32x4;

// ================= WMMA fragment plumbing (CDNA5 16x16x32 bf16) ===================
// A (16x32 MxK): lane&15 = M; lane>>4 picks K subgroup; 16 elements are two
// contiguous 8-element runs in a row-major LDS tile -> two ds_load_b128.
__device__ __forceinline__ bf16x16 load_a_frag(const __bf16* a, int lda, int lane) {
  bf16x16 fr;
  const int m  = lane & 15;
  const int kh = (lane >> 4) << 3;
#pragma unroll
  for (int i = 0; i < 16; ++i) {
    const int k = ((i >> 3) << 4) + kh + (i & 7);
    fr[i] = a[m * lda + k];
  }
  return fr;
}

// B tiles staged in LDS pre-swizzled into fragment order: block t (32Kx16N):
//   off = t*512 + lane*16 + i,  lane = (n&15) + 16*((k>>3)&1),  i = 8*(k>>4)+(k&7)
// -> a lane's fragment is 32 contiguous bytes: two ds_load_b128.
__device__ __forceinline__ bf16x16 load_b_frag_sw(const __bf16* bs, int t, int lane) {
  const __bf16* p = bs + t * 512 + lane * 16;
  const bf16x8 lo = *(const bf16x8*)p;
  const bf16x8 hi = *(const bf16x8*)(p + 8);
  bf16x16 fr;
#pragma unroll
  for (int j = 0; j < 8; ++j) { fr[j] = lo[j]; fr[8 + j] = hi[j]; }
  return fr;
}

// 4 B-fragment loads + 4 WMMAs; ds loads for the next fragment overlap with the
// previous WMMA executing in the XDL pipe.
__device__ __forceinline__ void wmma4(const __bf16* bs, int tb, int lane,
                                      const bf16x16& af, f32x8 acc[4]) {
  const bf16x16 b0 = load_b_frag_sw(bs, tb + 0, lane);
  const bf16x16 b1 = load_b_frag_sw(bs, tb + 1, lane);
  const bf16x16 b2 = load_b_frag_sw(bs, tb + 2, lane);
  const bf16x16 b3 = load_b_frag_sw(bs, tb + 3, lane);
  acc[0] = __builtin_amdgcn_wmma_f32_16x16x32_bf16(false, af, false, b0, (short)0, acc[0], false, false);
  acc[1] = __builtin_amdgcn_wmma_f32_16x16x32_bf16(false, af, false, b1, (short)0, acc[1], false, false);
  acc[2] = __builtin_amdgcn_wmma_f32_16x16x32_bf16(false, af, false, b2, (short)0, acc[2], false, false);
  acc[3] = __builtin_amdgcn_wmma_f32_16x16x32_bf16(false, af, false, b3, (short)0, acc[3], false, false);
}

// Staging helpers: LDS destinations are passed as parameters derived directly from
// the __shared__ arrays at the call site so addrspace(3) is preserved (ds_store_*).
__device__ __forceinline__ void stage_rowmajor(const __bf16* ap, const __bf16* bp,
                                               size_t ldb, __bf16* ad, __bf16* bd) {
  *(bf16x8*)ad = *(const bf16x8*)ap;
  const bf16x8 v0 = *(const bf16x8*)bp;
  const bf16x8 v1 = *(const bf16x8*)(bp + ldb);
#pragma unroll
  for (int j = 0; j < 8; ++j) {
    bf16x2 pr; pr[0] = v0[j]; pr[1] = v1[j];
    *(bf16x2*)(bd + j * 16) = pr;
  }
}

__device__ __forceinline__ void stage_rowmajor_f32a(const float* ap, const __bf16* bp,
                                                    size_t ldb, __bf16* ad, __bf16* bd) {
  const f32x4 u0 = *(const f32x4*)ap;
  const f32x4 u1 = *(const f32x4*)(ap + 4);
  bf16x8 w;
#pragma unroll
  for (int j = 0; j < 4; ++j) { w[j] = (__bf16)u0[j]; w[4 + j] = (__bf16)u1[j]; }
  *(bf16x8*)ad = w;
  const bf16x8 v0 = *(const bf16x8*)bp;
  const bf16x8 v1 = *(const bf16x8*)(bp + ldb);
#pragma unroll
  for (int j = 0; j < 8; ++j) {
    bf16x2 pr; pr[0] = v0[j]; pr[1] = v1[j];
    *(bf16x2*)(bd + j * 16) = pr;
  }
}

// B = K^T staging: thread holds 16 contiguous k for one key row n; packs 8 dwords.
__device__ __forceinline__ void stage_transposed(const __bf16* ap, const __bf16* kp,
                                                 int tkc, __bf16* ad, __bf16* bd) {
  *(bf16x8*)ad = *(const bf16x8*)ap;
  const bf16x8 v0 = *(const bf16x8*)kp;
  const bf16x8 v1 = *(const bf16x8*)(kp + 8);
  __bf16 tmp[16];
#pragma unroll
  for (int j = 0; j < 8; ++j) { tmp[j] = v0[j]; tmp[8 + j] = v1[j]; }
#pragma unroll
  for (int kk = 0; kk < 16; kk += 2) {
    const int k = tkc + kk;
    const int off = ((((k >> 3) & 1) << 4) * 16) + ((k >> 4) << 3) + (k & 7);
    bf16x2 pr; pr[0] = tmp[kk]; pr[1] = tmp[kk + 1];
    *(bf16x2*)(bd + off) = pr;
  }
}

// ================= Generic bf16 GEMM: C[M,N] f32 = A[M,K] * B[K,N] ================
__global__ __launch_bounds__(256) void gemm_bf16(const __bf16* __restrict__ A,
                                                 const __bf16* __restrict__ B,
                                                 float* __restrict__ C,
                                                 int M, int N, int K) {
  __shared__ __align__(16) __bf16 As[2][BM * BK];
  __shared__ __align__(16) __bf16 Bs[2][BK * BN];
  const int tid  = threadIdx.x;
  const int lane = tid & 31;
  const int wave = tid >> 5;
  const int wm = (wave & 3) << 4;   // 0,16,32,48
  const int wn = (wave >> 2) << 6;  // 0,64
  const int m0 = blockIdx.x * BM;
  const int n0 = blockIdx.y * BN;

  const int ar = tid >> 2;              // A row 0..63
  const int ac = (tid & 3) << 3;        // A col 0,8,16,24
  const int aoff = ar * BK + ac;
  const __bf16* aptr = A + (size_t)(m0 + ar) * K + ac;

  const int bk = (tid >> 4) << 1;       // 0,2,...,30
  const int bc = (tid & 15) << 3;       // 0,8,...,120
  const __bf16* bptr = B + (size_t)bk * N + n0 + bc;
  const size_t bstep = (size_t)BK * N;
  const int boff = (bc >> 4) * 512 + ((bc & 15) + (((bk >> 3) & 1) << 4)) * 16
                 + ((bk >> 4) << 3) + (bk & 7);

  f32x8 acc[4] = {};
  stage_rowmajor(aptr, bptr, N, &As[0][aoff], &Bs[0][boff]);
  const int nk = K / BK;
  for (int step = 0; step < nk; ++step) {
    __syncthreads();
    if (step + 1 < nk) {
      const int nb = (step + 1) & 1;
      stage_rowmajor(aptr + BK, bptr + bstep, N, &As[nb][aoff], &Bs[nb][boff]);
    }
    if (step + 2 < nk) {
      __builtin_prefetch(aptr + 2 * BK, 0, 3);
      __builtin_prefetch(bptr + 2 * bstep, 0, 3);
    }
    aptr += BK; bptr += bstep;
    const __bf16* as = As[step & 1];
    const __bf16* bs = Bs[step & 1];
    const bf16x16 af = load_a_frag(as + wm * BK, BK, lane);
    wmma4(bs, wn >> 4, lane, af, acc);
  }
#pragma unroll
  for (int t = 0; t < 4; ++t)
#pragma unroll
    for (int i = 0; i < 8; ++i) {
      const int m = m0 + wm + ((lane >> 4) << 3) + i;
      const int n = n0 + wn + t * 16 + (lane & 15);
      C[(size_t)m * N + n] = acc[t][i];
    }
}

// ========== Attention scores: logits = Q @ K^T * scale + causal mask ==============
__global__ __launch_bounds__(256) void attn_scores(const __bf16* __restrict__ Q,
                                                   const __bf16* __restrict__ Kc,
                                                   float* __restrict__ attn) {
  __shared__ __align__(16) __bf16 As[2][BM * BK];
  __shared__ __align__(16) __bf16 Bs[2][BK * BN];
  const int tid  = threadIdx.x;
  const int lane = tid & 31;
  const int wave = tid >> 5;
  const int wm = (wave & 3) << 4;
  const int wn = (wave >> 2) << 6;
  const int m0 = blockIdx.x * BM;
  const int n0 = blockIdx.y * BN;
  const int bh = blockIdx.z;
  const int b  = bh >> 3;
  const __bf16* Aq = Q  + (size_t)bh * SEQ * HEADD;
  const __bf16* Kb = Kc + (size_t)b  * SEQ * HEADD;
  float* Cd = attn + (size_t)bh * SEQ * SEQ;

  const int ar = tid >> 2;
  const int ac = (tid & 3) << 3;
  const int aoff = ar * BK + ac;
  const __bf16* aptr = Aq + (size_t)(m0 + ar) * HEADD + ac;

  const int tn  = tid & 127;
  const int tkc = (tid >> 7) << 4;      // 0 or 16
  const __bf16* kptr = Kb + (size_t)(n0 + tn) * HEADD + tkc;
  const int tbase = (tn >> 4) * 512 + (tn & 15) * 16;

  f32x8 acc[4] = {};
  stage_transposed(aptr, kptr, tkc, &As[0][aoff], &Bs[0][tbase]);
  const int nk = HEADD / BK;   // 8
  for (int step = 0; step < nk; ++step) {
    __syncthreads();
    if (step + 1 < nk) {
      const int nb = (step + 1) & 1;
      stage_transposed(aptr + BK, kptr + BK, tkc, &As[nb][aoff], &Bs[nb][tbase]);
    }
    aptr += BK; kptr += BK;
    const __bf16* as = As[step & 1];
    const __bf16* bs = Bs[step & 1];
    const bf16x16 af = load_a_frag(as + wm * BK, BK, lane);
    wmma4(bs, wn >> 4, lane, af, acc);
  }
#pragma unroll
  for (int t = 0; t < 4; ++t)
#pragma unroll
    for (int i = 0; i < 8; ++i) {
      const int m = m0 + wm + ((lane >> 4) << 3) + i;
      const int n = n0 + wn + t * 16 + (lane & 15);
      Cd[(size_t)m * SEQ + n] = acc[t][i] * SCALE_Q + (n > m ? NEGMASK : 0.0f);
    }
}

// ============ Adaptive-temperature softmax, one block per row, in place ===========
__global__ __launch_bounds__(256) void adaptive_softmax(float* __restrict__ attn) {
  __shared__ float row[SEQ];
  __shared__ float red[256];
  __shared__ float sMax, sSum, sBeta, sDen;
  float* p = attn + (size_t)blockIdx.x * SEQ;
  const int tid = threadIdx.x;
  for (int j = tid; j < SEQ; j += 256) row[j] = p[j];
  __syncthreads();
  float m = -3.4e38f;
  for (int j = tid; j < SEQ; j += 256) m = fmaxf(m, row[j]);
  red[tid] = m; __syncthreads();
  for (int s = 128; s > 0; s >>= 1) { if (tid < s) red[tid] = fmaxf(red[tid], red[tid + s]); __syncthreads(); }
  if (tid == 0) sMax = red[0];
  __syncthreads();
  const float mx = sMax;
  float se = 0.f, sl = 0.f;
  for (int j = tid; j < SEQ; j += 256) {
    const float e = __expf(row[j] - mx);
    se += e; sl += row[j] * e;
  }
  red[tid] = se; __syncthreads();
  for (int s = 128; s > 0; s >>= 1) { if (tid < s) red[tid] += red[tid + s]; __syncthreads(); }
  if (tid == 0) sSum = red[0];
  __syncthreads();
  red[tid] = sl; __syncthreads();
  for (int s = 128; s > 0; s >>= 1) { if (tid < s) red[tid] += red[tid + s]; __syncthreads(); }
  if (tid == 0) {
    const float lse  = mx + __logf(sSum);
    const float mean = red[0] / sSum;
    const float ent  = lse - mean;
    float beta = 1.0f;
    if (ent > 0.5f) {
      const float bp = ((((-0.037f * ent + 0.481f) * ent - 2.300f) * ent + 4.917f) * ent - 1.791f);
      beta = fmaxf(bp, 1.0f);
    }
    sBeta = beta;
  }
  __syncthreads();
  const float beta = sBeta;
  float s2 = 0.f;
  for (int j = tid; j < SEQ; j += 256) s2 += __expf(beta * (row[j] - mx));
  red[tid] = s2; __syncthreads();
  for (int s = 128; s > 0; s >>= 1) { if (tid < s) red[tid] += red[tid + s]; __syncthreads(); }
  if (tid == 0) sDen = red[0];
  __syncthreads();
  const float inv = 1.0f / sDen;
  for (int j = tid; j < SEQ; j += 256) p[j] = __expf(beta * (row[j] - mx)) * inv;
}

// ============== ctx = P(f32, converted on the fly) @ V(bf16) ======================
__global__ __launch_bounds__(256) void attn_v(const float* __restrict__ P,
                                              const __bf16* __restrict__ V,
                                              float* __restrict__ Ctx) {
  __shared__ __align__(16) __bf16 As[2][BM * BK];
  __shared__ __align__(16) __bf16 Bs[2][BK * BN];
  const int tid  = threadIdx.x;
  const int lane = tid & 31;
  const int wave = tid >> 5;
  const int wm = (wave & 3) << 4;
  const int wn = (wave >> 2) << 6;
  const int m0 = blockIdx.x * BM;
  const int n0 = blockIdx.y * BN;
  const int bh = blockIdx.z;
  const int b  = bh >> 3;
  const float*  Ap = P + (size_t)bh * SEQ * SEQ;
  const __bf16* Vb = V + (size_t)b  * SEQ * HEADD;
  float* Cd = Ctx + (size_t)bh * SEQ * HEADD;

  const int ar = tid >> 2;
  const int ac = (tid & 3) << 3;
  const int aoff = ar * BK + ac;
  const float* aptr = Ap + (size_t)(m0 + ar) * SEQ + ac;

  const int bk = (tid >> 4) << 1;
  const int bc = (tid & 15) << 3;
  const __bf16* bptr = Vb + (size_t)bk * HEADD + n0 + bc;
  const size_t bstep = (size_t)BK * HEADD;
  const int boff = (bc >> 4) * 512 + ((bc & 15) + (((bk >> 3) & 1) << 4)) * 16
                 + ((bk >> 4) << 3) + (bk & 7);

  f32x8 acc[4] = {};
  stage_rowmajor_f32a(aptr, bptr, HEADD, &As[0][aoff], &Bs[0][boff]);
  const int nk = SEQ / BK;   // 64
  for (int step = 0; step < nk; ++step) {
    __syncthreads();
    if (step + 1 < nk) {
      const int nb = (step + 1) & 1;
      stage_rowmajor_f32a(aptr + BK, bptr + bstep, HEADD, &As[nb][aoff], &Bs[nb][boff]);
    }
    if (step + 2 < nk) {
      __builtin_prefetch(aptr + 2 * BK, 0, 3);
      __builtin_prefetch(bptr + 2 * bstep, 0, 3);
    }
    aptr += BK; bptr += bstep;
    const __bf16* as = As[step & 1];
    const __bf16* bs = Bs[step & 1];
    const bf16x16 af = load_a_frag(as + wm * BK, BK, lane);
    wmma4(bs, wn >> 4, lane, af, acc);
  }
#pragma unroll
  for (int t = 0; t < 4; ++t)
#pragma unroll
    for (int i = 0; i < 8; ++i) {
      const int m = m0 + wm + ((lane >> 4) << 3) + i;
      const int n = n0 + wn + t * 16 + (lane & 15);
      Cd[(size_t)m * HEADD + n] = acc[t][i];
    }
}

// ================= Elementwise helpers ============================================
__global__ void pack_bf16(const float* __restrict__ src, __bf16* __restrict__ dst, int n) {
  const int i = blockIdx.x * 256 + threadIdx.x;
  if (i < n) dst[i] = (__bf16)src[i];
}

// dst[k*N + n] = src[n*K + k]  (weight row-major [N,K] -> B-matrix [K,N], bf16)
__global__ void pack_wT(const float* __restrict__ src, __bf16* __restrict__ dst, int N, int K) {
  const int i = blockIdx.x * 256 + threadIdx.x;
  if (i < N * K) {
    const int n = i % N;
    const int k = i / N;
    dst[i] = (__bf16)src[(size_t)n * K + k];
  }
}

// RoPE on Q + permute [B,S,NH,HD] -> [B,NH,S,HD], fp32 -> bf16
__global__ void rope_q(const float* __restrict__ Qf, __bf16* __restrict__ Qbf) {
  const int i = blockIdx.x * 256 + threadIdx.x;     // B*NH*SEQ*(HD/2)
  const int d = i & 127;
  const int s = (i >> 7) & (SEQ - 1);
  const int h = (i >> 18) & (NHEADS - 1);
  const int b = i >> 21;
  const float* src = Qf + (size_t)(b * SEQ + s) * (NHEADS * HEADD) + h * HEADD;
  const float x1 = src[d], x2 = src[d + 128];
  const float invf = __expf(-(float)d * (1.0f / 128.0f) * LN10000);
  float sn, cs; __sincosf((float)s * invf, &sn, &cs);
  __bf16* dst = Qbf + ((size_t)(b * NHEADS + h) * SEQ + s) * HEADD;
  dst[d]       = (__bf16)(x1 * cs - x2 * sn);
  dst[d + 128] = (__bf16)(x2 * cs + x1 * sn);
}

// RoPE on K (single KV head), fp32 -> bf16
__global__ void rope_k(const float* __restrict__ Kf, __bf16* __restrict__ Kbf) {
  const int i = blockIdx.x * 256 + threadIdx.x;     // B*SEQ*(HD/2)
  const int d = i & 127;
  const int s = (i >> 7) & (SEQ - 1);
  const int b = i >> 18;
  const float* src = Kf + (size_t)(b * SEQ + s) * HEADD;
  const float x1 = src[d], x2 = src[d + 128];
  const float invf = __expf(-(float)d * (1.0f / 128.0f) * LN10000);
  float sn, cs; __sincosf((float)s * invf, &sn, &cs);
  __bf16* dst = Kbf + (size_t)(b * SEQ + s) * HEADD;
  dst[d]       = (__bf16)(x1 * cs - x2 * sn);
  dst[d + 128] = (__bf16)(x2 * cs + x1 * sn);
}

// [B,NH,S,HD] fp32 ctx -> [B*S, NH*HD] bf16 for the output projection
__global__ void reshape_ctx(const float* __restrict__ Ctx, __bf16* __restrict__ Cbf) {
  const size_t i = (size_t)blockIdx.x * 256 + threadIdx.x;  // MROWS*HDIM
  const int col = (int)(i & (HDIM - 1));
  const int rowi = (int)(i >> 11);
  const int b = rowi >> 11;
  const int s = rowi & (SEQ - 1);
  const int h = col >> 8;
  const int d = col & (HEADD - 1);
  Cbf[i] = (__bf16)Ctx[((size_t)(b * NHEADS + h) * SEQ + s) * HEADD + d];
}

// ================= Host launcher ==================================================
extern "C" void kernel_launch(void* const* d_in, const int* in_sizes, int n_in,
                              void* d_out, int out_size, void* d_ws, size_t ws_size,
                              hipStream_t stream) {
  (void)in_sizes; (void)n_in; (void)out_size; (void)ws_size;
  const float* hs = (const float*)d_in[0];
  const float* wq = (const float*)d_in[1];
  const float* wk = (const float*)d_in[2];
  const float* wv = (const float*)d_in[3];
  const float* wo = (const float*)d_in[4];

  float* out  = (float*)d_out;                        // [B,S,H]
  float* attn = out + (size_t)MROWS * HDIM;           // [B,NH,S,S] (logits -> probs)

  char* w = (char*)d_ws;
  auto alloc = [&](size_t bytes) -> char* {
    char* p = w; w += (bytes + 255) & ~(size_t)255; return p;
  };
  __bf16* Xbf = (__bf16*)alloc((size_t)MROWS * HDIM * 2);
  __bf16* Wqb = (__bf16*)alloc((size_t)HDIM * HDIM * 2);
  __bf16* Wkb = (__bf16*)alloc((size_t)HDIM * HEADD * 2);
  __bf16* Wvb = (__bf16*)alloc((size_t)HDIM * HEADD * 2);
  __bf16* Wob = (__bf16*)alloc((size_t)HDIM * HDIM * 2);
  float*  Qf  = (float*)alloc((size_t)MROWS * HDIM * 4);
  float*  Kf  = (float*)alloc((size_t)MROWS * HEADD * 4);
  float*  Vf  = (float*)alloc((size_t)MROWS * HEADD * 4);
  __bf16* Qbf = (__bf16*)alloc((size_t)BATCH * NHEADS * SEQ * HEADD * 2);
  __bf16* Kbf = (__bf16*)alloc((size_t)BATCH * SEQ * HEADD * 2);
  __bf16* Vbf = (__bf16*)alloc((size_t)BATCH * SEQ * HEADD * 2);
  float*  Ctx = (float*)alloc((size_t)BATCH * NHEADS * SEQ * HEADD * 4);
  __bf16* Cbf = (__bf16*)alloc((size_t)MROWS * HDIM * 2);

  // 1) convert inputs / weights to bf16
  pack_bf16<<<(MROWS * HDIM) / 256, 256, 0, stream>>>(hs, Xbf, MROWS * HDIM);
  pack_wT<<<(HDIM * HDIM) / 256, 256, 0, stream>>>(wq, Wqb, HDIM, HDIM);
  pack_wT<<<(HEADD * HDIM) / 256, 256, 0, stream>>>(wk, Wkb, HEADD, HDIM);
  pack_wT<<<(HEADD * HDIM) / 256, 256, 0, stream>>>(wv, Wvb, HEADD, HDIM);
  pack_wT<<<(HDIM * HDIM) / 256, 256, 0, stream>>>(wo, Wob, HDIM, HDIM);

  // 2) QKV projections (WMMA)
  dim3 gq(MROWS / BM, HDIM / BN, 1);
  gemm_bf16<<<gq, 256, 0, stream>>>(Xbf, Wqb, Qf, MROWS, HDIM, HDIM);
  dim3 gkv(MROWS / BM, HEADD / BN, 1);
  gemm_bf16<<<gkv, 256, 0, stream>>>(Xbf, Wkb, Kf, MROWS, HEADD, HDIM);
  gemm_bf16<<<gkv, 256, 0, stream>>>(Xbf, Wvb, Vf, MROWS, HEADD, HDIM);

  // 3) RoPE + layout + bf16
  rope_q<<<(BATCH * NHEADS * SEQ * (HEADD / 2)) / 256, 256, 0, stream>>>(Qf, Qbf);
  rope_k<<<(BATCH * SEQ * (HEADD / 2)) / 256, 256, 0, stream>>>(Kf, Kbf);
  pack_bf16<<<(BATCH * SEQ * HEADD) / 256, 256, 0, stream>>>(Vf, Vbf, BATCH * SEQ * HEADD);

  // 4) scores (WMMA) -> logits in d_out's attn region
  dim3 gs(SEQ / BM, SEQ / BN, BATCH * NHEADS);
  attn_scores<<<gs, 256, 0, stream>>>(Qbf, Kbf, attn);

  // 5) adaptive-temperature softmax in place
  adaptive_softmax<<<BATCH * NHEADS * SEQ, 256, 0, stream>>>(attn);

  // 6) ctx = P @ V (WMMA)
  dim3 gv(SEQ / BM, HEADD / BN, BATCH * NHEADS);
  attn_v<<<gv, 256, 0, stream>>>(attn, Vbf, Ctx);

  // 7) output projection (WMMA) -> out
  reshape_ctx<<<(MROWS * HDIM) / 256, 256, 0, stream>>>(Ctx, Cbf);
  gemm_bf16<<<gq, 256, 0, stream>>>(Cbf, Wob, out, MROWS, HDIM, HDIM);
}
